// IHBR_17154099380775
// MI455X (gfx1250) — matrix operand order repaired and checked by reading, not compiled, not run on hardware
//
#include <hip/hip_runtime.h>
#include <math.h>

// ---------------------------------------------------------------------------
// IHBR live-path implementation for gfx1250 (MI455X), wave32 + WMMA bf16.
// Big GEMMs: C[M x 64] = epilogue(A[M x K] @ X[K x 64]) via
// v_wmma_f32_16x16x32_bf16. A is loaded f32 and converted in-register; X is
// pre-transposed/converted once into Xt[64][Kpad] bf16 (Kpad = ceil32(K)).
// ---------------------------------------------------------------------------

typedef __bf16 bf16_t;
typedef bf16_t v16bf __attribute__((ext_vector_type(16)));
typedef bf16_t v8bf  __attribute__((ext_vector_type(8)));
typedef float  v8f   __attribute__((ext_vector_type(8)));

#define EMB 64

// ---- WMMA operand loaders --------------------------------------------------
// A tile 16x32 bf16 (M x K): lanes 0-15 -> M=lane, K runs [k0..k0+7],[k0+16..23]
//                            lanes 16-31 -> M=lane-16, runs shifted by +8.
__device__ __forceinline__ v16bf load_a_full(const float* __restrict__ rp, int kbase) {
  const float4* p0 = (const float4*)(rp + kbase);
  const float4* p1 = (const float4*)(rp + kbase + 16);
  float4 f0 = p0[0], f1 = p0[1], f2 = p1[0], f3 = p1[1];
  v16bf a;
  a[0]=(bf16_t)f0.x;  a[1]=(bf16_t)f0.y;  a[2]=(bf16_t)f0.z;  a[3]=(bf16_t)f0.w;
  a[4]=(bf16_t)f1.x;  a[5]=(bf16_t)f1.y;  a[6]=(bf16_t)f1.z;  a[7]=(bf16_t)f1.w;
  a[8]=(bf16_t)f2.x;  a[9]=(bf16_t)f2.y;  a[10]=(bf16_t)f2.z; a[11]=(bf16_t)f2.w;
  a[12]=(bf16_t)f3.x; a[13]=(bf16_t)f3.y; a[14]=(bf16_t)f3.z; a[15]=(bf16_t)f3.w;
  return a;
}

__device__ __forceinline__ v16bf load_a_tail(const float* __restrict__ rp, int kbase, int K) {
  v16bf a;
#pragma unroll
  for (int i = 0; i < 8; ++i) {
    int k1 = kbase + i, k2 = kbase + 16 + i;
    a[i]     = (bf16_t)((k1 < K) ? rp[k1] : 0.f);
    a[8 + i] = (bf16_t)((k2 < K) ? rp[k2] : 0.f);
  }
  return a;
}

// B tile 32x16 bf16 (K x N) from Xt[64][kpad]: lane 0-15 -> N=n0+lane, K=[k0..k0+7],[k0+16..23]
//                                              lane 16-31 -> N=n0+lane-16, runs shifted by +8
__device__ __forceinline__ v16bf load_b(const bf16_t* __restrict__ Xt, int n0, int kpad, int k0) {
  int lane = threadIdx.x & 31;
  const bf16_t* p = Xt + (size_t)(n0 + (lane & 15)) * kpad + k0 + ((lane >> 4) << 3);
  const v8bf* vp = (const v8bf*)p;
  v8bf lo = vp[0], hi = vp[2];
  v16bf b;
#pragma unroll
  for (int i = 0; i < 8; ++i) { b[i] = lo[i]; b[8 + i] = hi[i]; }
  return b;
}

// One 16x16 f32 accumulator tile: rows [row0..row0+15], cols [n0..n0+15].
__device__ __forceinline__ v8f gemm_tile(const float* __restrict__ A,
                                         const bf16_t* __restrict__ Xt,
                                         int M, int K, int kpad, int row0, int n0) {
  int lane = threadIdx.x & 31;
  int m = row0 + (lane & 15);
  if (m >= M) m = M - 1;                 // clamp: keeps EXEC all-ones, stores guarded later
  const float* rp = A + (size_t)m * K;
  int koff = (lane >> 4) << 3;           // +8 K offset for upper half-wave
  v8f acc = {0.f,0.f,0.f,0.f,0.f,0.f,0.f,0.f};
  int kend = K & ~31;
  for (int k0 = 0; k0 < kend; k0 += 32) {
    v16bf a = load_a_full(rp, k0 + koff);
    v16bf b = load_b(Xt, n0, kpad, k0);
    acc = __builtin_amdgcn_wmma_f32_16x16x32_bf16(false, a, false, b, (short)0, acc, false, false);
  }
  if (kend < K) {
    v16bf a = load_a_tail(rp, kend + koff, K);
    v16bf b = load_b(Xt, n0, kpad, kend);  // Xt pad region is zero-filled
    acc = __builtin_amdgcn_wmma_f32_16x16x32_bf16(false, a, false, b, (short)0, acc, false, false);
  }
  return acc;
}

// C/D layout: element r of lane L -> C[row0 + r + 8*(L>>4)][n0 + (L&15)]
#define TILE_COORDS() \
  int lane = threadIdx.x & 31; \
  int wave = threadIdx.x >> 5; \
  int row0 = blockIdx.x * 32 + ((wave >> 2) << 4); \
  int n0 = (wave & 3) << 4; \
  (void)lane

// ---- GEMM kernels with fused epilogues ------------------------------------

// items0 = relu(A_i@F) + relu(B_i@F) + F
__global__ void k_items0(const float* __restrict__ Ai, const float* __restrict__ Bi,
                         const bf16_t* __restrict__ Xt, const float* __restrict__ feat,
                         float* __restrict__ C, int M, int K, int kpad) {
  TILE_COORDS();
  v8f a1 = gemm_tile(Ai, Xt, M, K, kpad, row0, n0);
  v8f a2 = gemm_tile(Bi, Xt, M, K, kpad, row0, n0);
  int n = n0 + (lane & 15);
  int mb = row0 + ((lane >> 4) << 3);
#pragma unroll
  for (int r = 0; r < 8; ++r) {
    int m = mb + r;
    if (m < M)
      C[(size_t)m * EMB + n] = fmaxf(a1[r], 0.f) + fmaxf(a2[r], 0.f) + feat[(size_t)m * EMB + n];
  }
}

// C = A@X (q/k/v/out projections, h = f_ub0 @ W_ub)
__global__ void k_plain(const float* __restrict__ A, const bf16_t* __restrict__ Xt,
                        float* __restrict__ C, int M, int K, int kpad) {
  TILE_COORDS();
  v8f acc = gemm_tile(A, Xt, M, K, kpad, row0, n0);
  int n = n0 + (lane & 15);
  int mb = row0 + ((lane >> 4) << 3);
#pragma unroll
  for (int r = 0; r < 8; ++r) {
    int m = mb + r;
    if (m < M) C[(size_t)m * EMB + n] = acc[r];
  }
}

// rb = relu(bi_avg@items); bundles_f = rb + bundles_feature
__global__ void k_bundles(const float* __restrict__ A, const bf16_t* __restrict__ Xt,
                          const float* __restrict__ bfeat, float* __restrict__ rb,
                          float* __restrict__ bundles_f, int M, int K, int kpad) {
  TILE_COORDS();
  v8f acc = gemm_tile(A, Xt, M, K, kpad, row0, n0);
  int n = n0 + (lane & 15);
  int mb = row0 + ((lane >> 4) << 3);
#pragma unroll
  for (int r = 0; r < 8; ++r) {
    int m = mb + r;
    if (m < M) {
      float v = fmaxf(acc[r], 0.f);
      rb[(size_t)m * EMB + n] = v;
      bundles_f[(size_t)m * EMB + n] = v + bfeat[(size_t)m * EMB + n];
    }
  }
}

// users_f = relu(ui_avg@items) + relu(ub_avg@bundles_f) + users_feature
__global__ void k_users(const float* __restrict__ A1, const bf16_t* __restrict__ Xt1,
                        int K1, int kpad1,
                        const float* __restrict__ A2, const bf16_t* __restrict__ Xt2,
                        int K2, int kpad2,
                        const float* __restrict__ ufeat, float* __restrict__ C, int M) {
  TILE_COORDS();
  v8f a1 = gemm_tile(A1, Xt1, M, K1, kpad1, row0, n0);
  v8f a2 = gemm_tile(A2, Xt2, M, K2, kpad2, row0, n0);
  int n = n0 + (lane & 15);
  int mb = row0 + ((lane >> 4) << 3);
#pragma unroll
  for (int r = 0; r < 8; ++r) {
    int m = mb + r;
    if (m < M)
      C[(size_t)m * EMB + n] = fmaxf(a1[r], 0.f) + fmaxf(a2[r], 0.f) + ufeat[(size_t)m * EMB + n];
  }
}

// f_ub = relu(ub_graph @ h + b_ub)
__global__ void k_relu_bias(const float* __restrict__ A, const bf16_t* __restrict__ Xt,
                            const float* __restrict__ bias, float* __restrict__ C,
                            int M, int K, int kpad) {
  TILE_COORDS();
  v8f acc = gemm_tile(A, Xt, M, K, kpad, row0, n0);
  int n = n0 + (lane & 15);
  float bn = bias[n];
  int mb = row0 + ((lane >> 4) << 3);
#pragma unroll
  for (int r = 0; r < 8; ++r) {
    int m = mb + r;
    if (m < M) C[(size_t)m * EMB + n] = fmaxf(acc[r] + bn, 0.f);
  }
}

// ---- X (K x 64 f32) -> Xt (64 x Kpad bf16), zero-padded -------------------
__global__ void k_xt(const float* __restrict__ X, bf16_t* __restrict__ Xt, int K, int kpad) {
  int n = blockIdx.y;
  int k = blockIdx.x * 256 + threadIdx.x;
  if (k < kpad)
    Xt[(size_t)n * kpad + k] = (k < K) ? (bf16_t)X[(size_t)k * EMB + n] : (bf16_t)0.f;
}

// ---- Flash attention: 8 heads, dh=8, N tokens -----------------------------
// Block: 32 queries x 8 threads/query. K/V chunks of 512 staged in LDS (32KB).
__global__ void k_mha(const float* __restrict__ q, const float* __restrict__ k,
                      const float* __restrict__ v, float* __restrict__ o, int N) {
  __shared__ float Ks[512 * 8];
  __shared__ float Vs[512 * 8];
  int h = blockIdx.y;
  int qrow = blockIdx.x * 32 + (threadIdx.x >> 3);
  int j = threadIdx.x & 7;
  float qv[8];
#pragma unroll
  for (int d = 0; d < 8; ++d)
    qv[d] = q[(size_t)qrow * EMB + h * 8 + d] * 0.35355339059327373f; // 1/sqrt(8)
  float m = -1e30f, l = 0.f;
  float acc[8] = {0, 0, 0, 0, 0, 0, 0, 0};
  for (int c0 = 0; c0 < N; c0 += 512) {
    __syncthreads();
    for (int idx = threadIdx.x; idx < 512 * 8; idx += 256) {
      int key = idx >> 3, d = idx & 7;
      size_t g = (size_t)(c0 + key) * EMB + h * 8 + d;
      Ks[idx] = k[g];
      Vs[idx] = v[g];
    }
    __syncthreads();
    for (int key = j; key < 512; key += 8) {
      const float* kp = &Ks[key * 8];
      float s = qv[0]*kp[0] + qv[1]*kp[1] + qv[2]*kp[2] + qv[3]*kp[3]
              + qv[4]*kp[4] + qv[5]*kp[5] + qv[6]*kp[6] + qv[7]*kp[7];
      float mn = fmaxf(m, s);
      float corr = __expf(m - mn);
      float pr = __expf(s - mn);
      l = l * corr + pr;
      const float* vp = &Vs[key * 8];
#pragma unroll
      for (int d = 0; d < 8; ++d) acc[d] = acc[d] * corr + pr * vp[d];
      m = mn;
    }
  }
  // merge the 8 partial online-softmax states of this query
#pragma unroll
  for (int off = 1; off < 8; off <<= 1) {
    float mo = __shfl_xor(m, off, 8);
    float lo = __shfl_xor(l, off, 8);
    float mn = fmaxf(m, mo);
    float cs = __expf(m - mn), co = __expf(mo - mn);
    l = l * cs + lo * co;
#pragma unroll
    for (int d = 0; d < 8; ++d)
      acc[d] = acc[d] * cs + __shfl_xor(acc[d], off, 8) * co;
    m = mn;
  }
  o[(size_t)qrow * EMB + h * 8 + j] = acc[j] / l;
}

// ---- att2 fusion + l2norms -> all_ub[9000 x 192]; one wave per row --------
__global__ void k_fuse(const float* __restrict__ f_ub0, const float* __restrict__ rb,
                       const float* __restrict__ f_ub, const float* __restrict__ attw,
                       float* __restrict__ all_ub, int Uc, int NTc) {
  int row = (blockIdx.x * blockDim.x + threadIdx.x) >> 5;
  int lane = threadIdx.x & 31;
  if (row >= NTc) return;
  const float* f0p = f_ub0 + (size_t)row * EMB;
  float f0a = f0p[lane], f0b = f0p[32 + lane];
  float x2a, x2b;
  if (row < Uc) {                       // u1 == f_ub0[:U] exactly
    x2a = f0a; x2b = f0b;
  } else {                              // b1 = l2norm(rb)
    const float* rp = rb + (size_t)(row - Uc) * EMB;
    float ra = rp[lane], rbv = rp[32 + lane];
    float ss = ra * ra + rbv * rbv;
#pragma unroll
    for (int t = 16; t > 0; t >>= 1) ss += __shfl_xor(ss, t, 32);
    float inv = 1.f / fmaxf(sqrtf(ss), 1e-12f);
    x2a = ra * inv; x2b = rbv * inv;
  }
  float wa = attw[lane], wb = attw[32 + lane];
  float s0 = f0a * wa + f0b * wb;
  float s1 = x2a * wa + x2b * wb;
#pragma unroll
  for (int t = 16; t > 0; t >>= 1) { s0 += __shfl_xor(s0, t, 32); s1 += __shfl_xor(s1, t, 32); }
  float mx = fmaxf(s0, s1);
  float e0 = __expf(s0 - mx), e1 = __expf(s1 - mx);
  float inv = 1.f / (e0 + e1);
  float a0 = e0 * inv, a1 = e1 * inv;
  float fa = a0 * f0a + a1 * x2a;
  float fb = a0 * f0b + a1 * x2b;
  const float* gp = f_ub + (size_t)row * EMB;
  float ga = gp[lane], gb = gp[32 + lane];
  float ss2 = ga * ga + gb * gb + fa * fa + fb * fb;
#pragma unroll
  for (int t = 16; t > 0; t >>= 1) ss2 += __shfl_xor(ss2, t, 32);
  float inv2 = 1.f / fmaxf(sqrtf(ss2), 1e-12f);
  float* op = all_ub + (size_t)row * 192;
  op[lane] = f0a;           op[32 + lane] = f0b;
  op[64 + lane] = ga * inv2; op[96 + lane] = gb * inv2;
  op[128 + lane] = fa * inv2; op[160 + lane] = fb * inv2;
}

// ---- gather + MLP head -----------------------------------------------------
__global__ void k_final(const float* __restrict__ all_ub, const int* __restrict__ users,
                        const int* __restrict__ bundles,
                        const float* __restrict__ p1W, const float* __restrict__ p1b,
                        const float* __restrict__ p2W, const float* __restrict__ p2b,
                        float* __restrict__ out, int nb, int Uc) {
  __shared__ float W[576 * 8];
  for (int i = threadIdx.x; i < 576 * 8; i += 256) W[i] = p1W[i];
  __syncthreads();
  int i = blockIdx.x * 256 + threadIdx.x;
  if (i >= nb) return;
  const float* ue = all_ub + (size_t)users[i] * 192;
  const float* be = all_ub + (size_t)(Uc + bundles[i]) * 192;
  float hid[8];
#pragma unroll
  for (int j = 0; j < 8; ++j) hid[j] = p1b[j];
  for (int kk = 0; kk < 192; ++kk) {
    float u = ue[kk], b = be[kk];
    float ub = u * b;
#pragma unroll
    for (int j = 0; j < 8; ++j)
      hid[j] += ub * W[kk * 8 + j] + b * W[(kk + 192) * 8 + j] + u * W[(kk + 384) * 8 + j];
  }
  float res = p2b[0];
#pragma unroll
  for (int j = 0; j < 8; ++j) res += fmaxf(hid[j], 0.f) * p2W[j];
  out[i] = res;
}

// ---------------------------------------------------------------------------
extern "C" void kernel_launch(void* const* d_in, const int* in_sizes, int n_in,
                              void* d_out, int out_size, void* d_ws, size_t ws_size,
                              hipStream_t stream) {
  const int U = 6000, I = 3072, BN = 3000, NT = U + BN;
  const int KP_I = 3072, KP_B = 3008, KP_N = 9024;   // ceil32 pads

  const float* users_feature   = (const float*)d_in[0];
  const float* items_feature   = (const float*)d_in[1];
  const float* bundles_feature = (const float*)d_in[2];
  const float* Wq    = (const float*)d_in[3];
  const float* Wk    = (const float*)d_in[4];
  const float* Wv    = (const float*)d_in[5];
  const float* Wo    = (const float*)d_in[6];
  const float* W_ub  = (const float*)d_in[7];
  const float* b_ub  = (const float*)d_in[8];
  // d_in[9] W_ui, [10] b_ui, [12] att_i_w, [22] ib_avg, [24] ui_graph: dead code
  const float* att_b_w = (const float*)d_in[11];
  const float* p1W = (const float*)d_in[13];
  const float* p1b = (const float*)d_in[14];
  const float* p2W = (const float*)d_in[15];
  const float* p2b = (const float*)d_in[16];
  const float* A_i    = (const float*)d_in[17];
  const float* B_i    = (const float*)d_in[18];
  const float* bi_avg = (const float*)d_in[19];
  const float* ui_avg = (const float*)d_in[20];
  const float* ub_avg = (const float*)d_in[21];
  const float* ub_graph = (const float*)d_in[23];
  const int* users   = (const int*)d_in[25];
  const int* bundles = (const int*)d_in[26];
  const int batch = in_sizes[25];

  // workspace carving (256B aligned)
  char* wp = (char*)d_ws;
  auto carve = [&](size_t bytes) { void* r = (void*)wp; wp += (bytes + 255) & ~(size_t)255; return r; };
  float* items0 = (float*)carve((size_t)I * EMB * 4);
  float* qb     = (float*)carve((size_t)I * EMB * 4);
  float* kb     = (float*)carve((size_t)I * EMB * 4);
  float* vb     = (float*)carve((size_t)I * EMB * 4);
  float* attn   = (float*)carve((size_t)I * EMB * 4);
  float* itemsv = (float*)carve((size_t)I * EMB * 4);
  float* rb     = (float*)carve((size_t)BN * EMB * 4);
  float* f_ub0  = (float*)carve((size_t)NT * EMB * 4);   // [users_f ; bundles_f]
  float* hbuf   = (float*)carve((size_t)NT * EMB * 4);
  float* f_ub   = (float*)carve((size_t)NT * EMB * 4);
  float* all_ub = (float*)carve((size_t)NT * 192 * 4);
  bf16_t* Xt1  = (bf16_t*)carve((size_t)EMB * KP_I * 2);
  bf16_t* Xt2  = (bf16_t*)carve((size_t)EMB * KP_B * 2);
  bf16_t* Xt3  = (bf16_t*)carve((size_t)EMB * KP_N * 2);
  bf16_t* XtWq = (bf16_t*)carve((size_t)EMB * EMB * 2);
  bf16_t* XtWk = (bf16_t*)carve((size_t)EMB * EMB * 2);
  bf16_t* XtWv = (bf16_t*)carve((size_t)EMB * EMB * 2);
  bf16_t* XtWo = (bf16_t*)carve((size_t)EMB * EMB * 2);
  bf16_t* XtWub= (bf16_t*)carve((size_t)EMB * EMB * 2);
  float* bundles_f = f_ub0 + (size_t)U * EMB;

  dim3 blk(256);
  auto gtiles = [](int M) { return dim3((unsigned)((M + 31) / 32)); };
  auto gxt    = [](int kpad) { return dim3((unsigned)((kpad + 255) / 256), EMB); };

  // 1) items0 = relu(A_i@F)+relu(B_i@F)+F
  k_xt<<<gxt(KP_I), blk, 0, stream>>>(items_feature, Xt1, I, KP_I);
  k_items0<<<gtiles(I), blk, 0, stream>>>(A_i, B_i, Xt1, items_feature, items0, I, I, KP_I);

  // 2) multi-head attention
  k_xt<<<gxt(EMB), blk, 0, stream>>>(Wq, XtWq, EMB, EMB);
  k_xt<<<gxt(EMB), blk, 0, stream>>>(Wk, XtWk, EMB, EMB);
  k_xt<<<gxt(EMB), blk, 0, stream>>>(Wv, XtWv, EMB, EMB);
  k_xt<<<gxt(EMB), blk, 0, stream>>>(Wo, XtWo, EMB, EMB);
  k_xt<<<gxt(EMB), blk, 0, stream>>>(W_ub, XtWub, EMB, EMB);
  k_plain<<<gtiles(I), blk, 0, stream>>>(items0, XtWq, qb, I, EMB, EMB);
  k_plain<<<gtiles(I), blk, 0, stream>>>(items0, XtWk, kb, I, EMB, EMB);
  k_plain<<<gtiles(I), blk, 0, stream>>>(items0, XtWv, vb, I, EMB, EMB);
  k_mha<<<dim3((unsigned)(I / 32), 8), blk, 0, stream>>>(qb, kb, vb, attn, I);
  k_plain<<<gtiles(I), blk, 0, stream>>>(attn, XtWo, itemsv, I, EMB, EMB);

  // 3) bundles_f / rb
  k_xt<<<gxt(KP_I), blk, 0, stream>>>(itemsv, Xt1, I, KP_I);
  k_bundles<<<gtiles(BN), blk, 0, stream>>>(bi_avg, Xt1, bundles_feature, rb, bundles_f, BN, I, KP_I);

  // 4) users_f (two fused GEMMs)
  k_xt<<<gxt(KP_B), blk, 0, stream>>>(bundles_f, Xt2, BN, KP_B);
  k_users<<<gtiles(U), blk, 0, stream>>>(ui_avg, Xt1, I, KP_I, ub_avg, Xt2, BN, KP_B,
                                         users_feature, f_ub0, U);

  // 5) f_ub = relu(ub_graph @ (f_ub0 @ W_ub) + b_ub)
  k_plain<<<gtiles(NT), blk, 0, stream>>>(f_ub0, XtWub, hbuf, NT, EMB, EMB);
  k_xt<<<gxt(KP_N), blk, 0, stream>>>(hbuf, Xt3, NT, KP_N);
  k_relu_bias<<<gtiles(NT), blk, 0, stream>>>(ub_graph, Xt3, b_ub, f_ub, NT, NT, KP_N);

  // 6) att2 fusion + l2norms -> all_ub
  k_fuse<<<dim3((unsigned)((NT * 32 + 255) / 256)), blk, 0, stream>>>(f_ub0, rb, f_ub, att_b_w,
                                                                      all_ub, U, NT);

  // 7) gather + MLP head
  k_final<<<dim3((unsigned)((batch + 255) / 256)), blk, 0, stream>>>(
      all_ub, users, bundles, p1W, p1b, p2W, p2b, (float*)d_out, batch, U);

  (void)n_in; (void)out_size; (void)ws_size;
}
